// MultiScaleRetention_8985071583702
// MI455X (gfx1250) — compile-verified, matrix operand order checked by reading, not run on hardware
//
#include <hip/hip_runtime.h>

typedef __attribute__((ext_vector_type(16))) _Float16 v16h;
typedef __attribute__((ext_vector_type(8)))  _Float16 v8h;
typedef __attribute__((ext_vector_type(8)))  float    v8f;

#define DMODEL 512
#define NHEADS 8
#define DHEAD  64
#define TSEQ   2048
#define BT     4096   // B*T rows

__device__ __forceinline__ v16h cat8(v8h lo, v8h hi) {
  v16h r;
#pragma unroll
  for (int i = 0; i < 8; ++i) { r[i] = lo[i]; r[i + 8] = hi[i]; }
  return r;
}

__device__ __forceinline__ v8f wmma_f16(v16h a, v16h b, v8f c) {
  // D = A(16x32 f16) * B(32x16 f16) + C(16x16 f32)
  return __builtin_amdgcn_wmma_f32_16x16x32_f16(false, a, false, b, (short)0, c,
                                                false, false);
}

// ---------------------------------------------------------------------------
// Generic 4096x512 @ 512x512 GEMM, f32 inputs converted to f16 for WMMA.
// MODE 0: f32 row-major out (* scale)
// MODE 1: f16 row-major out (* scale)
// MODE 2: f16 transposed out  Out[col * BT + row]  (* scale)
// Block: 256 threads (8 waves). Tile: 64(M) x 64(N), K-step 32.
// ---------------------------------------------------------------------------
template <int MODE>
__global__ __launch_bounds__(256) void gemm512(const float* __restrict__ A,
                                               const float* __restrict__ W,
                                               void* __restrict__ Out,
                                               float scale) {
  __shared__ _Float16 As[64][40];   // As[m][k]
  __shared__ _Float16 Bst[64][40];  // Bst[n][k] = W[k0+k][n0+n]

  const int t    = threadIdx.x;
  const int lane = t & 31;
  const int w    = t >> 5;
  const int m0   = blockIdx.x * 64;
  const int n0   = blockIdx.y * 64;

  const int r0  = (w & 3) * 16;        // wave's 16-row block
  const int c0  = (w >> 2) * 32;       // wave's 32-col block (two 16-col tiles)
  const int kb  = (lane >> 4) * 8;     // per-lane K sub-chunk (A/B frag layout)
  const int l15 = lane & 15;

  v8f acc0 = {}, acc1 = {};

  for (int k0 = 0; k0 < DMODEL; k0 += 32) {
    __syncthreads();
    {
      // Stage A tile 64x32 (4 threads/row, 8 f32 each)
      const int row = t >> 2, cs = (t & 3) * 8;
      const float* ap = A + (size_t)(m0 + row) * DMODEL + k0 + cs;
#pragma unroll
      for (int j = 0; j < 8; ++j) As[row][cs + j] = (_Float16)ap[j];
      // Stage W tile 32x64 transposed (8 threads/row, 8 f32 each)
      const int krow = t >> 3, ns = (t & 7) * 8;
      const float* wp = W + (size_t)(k0 + krow) * DMODEL + n0 + ns;
#pragma unroll
      for (int j = 0; j < 8; ++j) Bst[ns + j][krow] = (_Float16)wp[j];
      if (k0 + 32 < DMODEL) {
        __builtin_prefetch(ap + 32, 0, 0);                 // global_prefetch_b8
        __builtin_prefetch(wp + (size_t)32 * DMODEL, 0, 0);
      }
    }
    __syncthreads();

    // A fragment (16x32 f16): lane holds row r0+l15, K chunks kb..kb+7, kb+16..kb+23
    const int mrow = r0 + l15;
    v16h af = cat8(*(const v8h*)&As[mrow][kb], *(const v8h*)&As[mrow][kb + 16]);
    // B fragments (32x16 f16): lane holds col, K chunks as above
    const int nc = c0 + l15;
    v16h b0 = cat8(*(const v8h*)&Bst[nc][kb], *(const v8h*)&Bst[nc][kb + 16]);
    v16h b1 = cat8(*(const v8h*)&Bst[nc + 16][kb], *(const v8h*)&Bst[nc + 16][kb + 16]);
    acc0 = wmma_f16(af, b0, acc0);
    acc1 = wmma_f16(af, b1, acc1);
  }

  // C layout: element v -> (row = v + 8*(lane>>4), col = lane&15)
  const int mrow = m0 + r0 + 8 * (lane >> 4);
  const int ncol = n0 + c0 + l15;
#pragma unroll
  for (int v = 0; v < 8; ++v) {
    const int gr = mrow + v;
    const float x0 = acc0[v] * scale;
    const float x1 = acc1[v] * scale;
    if (MODE == 0) {
      float* o = (float*)Out;
      o[(size_t)gr * DMODEL + ncol]      = x0;
      o[(size_t)gr * DMODEL + ncol + 16] = x1;
    } else if (MODE == 1) {
      _Float16* o = (_Float16*)Out;
      o[(size_t)gr * DMODEL + ncol]      = (_Float16)x0;
      o[(size_t)gr * DMODEL + ncol + 16] = (_Float16)x1;
    } else {
      _Float16* o = (_Float16*)Out;
      o[(size_t)ncol * BT + gr]        = (_Float16)x0;
      o[(size_t)(ncol + 16) * BT + gr] = (_Float16)x1;
    }
  }
}

// ---------------------------------------------------------------------------
// Retention: O[b,h,n,:] = sum_m gamma_h^(n-m) * (Q K^T)[n,m] * V[m,:]  (m<=n)
// Grid: (T/64, H, B). Block 128 = 4 waves; wave owns 16 query rows.
// Q pre-scaled by 1/sqrt(dh)=0.125 in the projection.
// Decay factored as gamma^(n-m) = rowfac(n,m0) * colfac(m-m0); rowfac updated
// multiplicatively per key block -> zero transcendentals in the hot loop.
// ---------------------------------------------------------------------------
__global__ __launch_bounds__(128) void retention_kernel(
    const _Float16* __restrict__ Qh,   // (B*T, 512) f16, pre-scaled
    const _Float16* __restrict__ Kh,   // (B*T, 512) f16
    const _Float16* __restrict__ Vt,   // (512, B*T) f16 (transposed V)
    float* __restrict__ Ret) {         // (B*T, 512) f32
  __shared__ _Float16 Pl[4][16][40];   // per-wave decayed-score tile (16x32)

  const int lane = threadIdx.x & 31;
  const int w    = threadIdx.x >> 5;
  const int h    = blockIdx.y;
  const int b    = blockIdx.z;
  const int q0   = blockIdx.x * 64 + w * 16;  // query base within batch

  const int kb   = (lane >> 4) * 8;
  const int l15  = lane & 15;
  const int mloc = 8 * (lane >> 4);

  const float gamma = 1.0f - __builtin_amdgcn_exp2f(-5.0f - (float)h);
  const float lg    = __builtin_log2f(gamma);

  // colfac = gamma^(-(key - m0)) for the two 16-col key tiles (exp in [0,32))
  const float cw0 = __builtin_amdgcn_exp2f(-lg * (float)l15);
  const float cw1 = __builtin_amdgcn_exp2f(-lg * (float)(16 + l15));
  // rowfac[v] = gamma^(qn - m0), starting at m0 = 0; step by gamma^-32
  const float stepf = __builtin_amdgcn_exp2f(-lg * 32.0f);
  float rf[8];
#pragma unroll
  for (int v = 0; v < 8; ++v)
    rf[v] = __builtin_amdgcn_exp2f(lg * (float)(q0 + mloc + v));

  // Q A-fragments for K-dim 64 (two 16x32 tiles), loaded once from global
  const v8h* qp = (const v8h*)(Qh + (size_t)(b * TSEQ + q0 + l15) * DMODEL +
                               h * DHEAD + kb);
  const v16h aq0 = cat8(qp[0], qp[2]);
  const v16h aq1 = cat8(qp[4], qp[6]);

  v8f oacc[4] = {{}, {}, {}, {}};

  for (int m0 = 0; m0 <= q0 + 15; m0 += 32) {   // 32-key blocks up to diagonal
#pragma unroll
    for (int tkey = 0; tkey < 2; ++tkey) {
      const int key = m0 + tkey * 16 + l15;     // this lane's key column
      const float cw = tkey ? cw1 : cw0;
      // K^T B-fragment: lane holds key row, feature chunks per B layout
      const v8h* kp = (const v8h*)(Kh + (size_t)(b * TSEQ + key) * DMODEL +
                                   h * DHEAD + kb);
      v16h bk0 = cat8(kp[0], kp[2]);
      v16h bk1 = cat8(kp[4], kp[6]);
      v8f s = {};
      s = wmma_f16(aq0, bk0, s);
      s = wmma_f16(aq1, bk1, s);
      // causal multi-scale decay on C fragment, convert to f16 A-tile in LDS
#pragma unroll
      for (int v = 0; v < 8; ++v) {
        const int qn = q0 + mloc + v;
        const float wgt = (qn >= key) ? rf[v] * cw : 0.0f;
        Pl[w][mloc + v][tkey * 16 + l15] = (_Float16)(s[v] * wgt);
      }
    }
#pragma unroll
    for (int v = 0; v < 8; ++v) rf[v] *= stepf;  // advance rowfac to next m0

    // LDS is in-order per wave; explicit wait keeps the RAW airtight
    asm volatile("s_wait_dscnt 0" ::: "memory");

    // P A-fragment (16x32)
    v16h pf = cat8(*(const v8h*)&Pl[w][l15][kb],
                   *(const v8h*)&Pl[w][l15][kb + 16]);
    // V B-fragments: 4 feature tiles of 16; contiguous thanks to Vt transpose
#pragma unroll
    for (int nt = 0; nt < 4; ++nt) {
      const int d = nt * 16 + l15;
      const v8h* vp = (const v8h*)(Vt + (size_t)(h * DHEAD + d) * BT +
                                   b * TSEQ + m0 + kb);
      v16h bv = cat8(vp[0], vp[2]);
      oacc[nt] = wmma_f16(pf, bv, oacc[nt]);
    }
  }

#pragma unroll
  for (int nt = 0; nt < 4; ++nt)
#pragma unroll
    for (int v = 0; v < 8; ++v)
      Ret[(size_t)(b * TSEQ + q0 + mloc + v) * DMODEL + h * DHEAD + nt * 16 +
          l15] = oacc[nt][v];
}

// ---------------------------------------------------------------------------
// GroupNorm(8, 512) per token + gate: Z = silu(Gp + gn(Ret)*w + b)
// One 64-thread block per (token, head) group.
// ---------------------------------------------------------------------------
__global__ __launch_bounds__(64) void gn_gate_kernel(
    const float* __restrict__ Ret, const float* __restrict__ Gp,
    const float* __restrict__ gw, const float* __restrict__ gb,
    float* __restrict__ Z) {
  __shared__ float s1[64], s2[64];
  const int d    = threadIdx.x;
  const int gidx = blockIdx.x;
  const int h    = gidx & (NHEADS - 1);
  const int tok  = gidx >> 3;
  const size_t base = (size_t)tok * DMODEL + h * DHEAD;
  const float x = Ret[base + d];
  s1[d] = x;
  s2[d] = x * x;
  __syncthreads();
#pragma unroll
  for (int off = 32; off > 0; off >>= 1) {
    if (d < off) { s1[d] += s1[d + off]; s2[d] += s2[d + off]; }
    __syncthreads();
  }
  const float mean = s1[0] * (1.0f / 64.0f);
  const float var  = s2[0] * (1.0f / 64.0f) - mean * mean;
  const float nrm  = (x - mean) * rsqrtf(var + 1e-5f);
  const float zz   = Gp[base + d] + nrm * gw[h * DHEAD + d] + gb[h * DHEAD + d];
  Z[base + d] = zz / (1.0f + __expf(-zz));
}

// ---------------------------------------------------------------------------
extern "C" void kernel_launch(void* const* d_in, const int* in_sizes, int n_in,
                              void* d_out, int out_size, void* d_ws,
                              size_t ws_size, hipStream_t stream) {
  const float* x  = (const float*)d_in[0];
  const float* WQ = (const float*)d_in[1];
  const float* WK = (const float*)d_in[2];
  const float* WV = (const float*)d_in[3];
  const float* WG = (const float*)d_in[4];
  const float* WO = (const float*)d_in[5];
  const float* gw = (const float*)d_in[6];
  const float* gb = (const float*)d_in[7];

  char* ws = (char*)d_ws;
  _Float16* Qh = (_Float16*)(ws);                  // 4 MB: Q * 0.125, f16
  _Float16* Kh = (_Float16*)(ws + (4u << 20));     // 4 MB
  _Float16* Vt = (_Float16*)(ws + (8u << 20));     // 4 MB (transposed)
  float*    Gp = (float*)(ws + (12u << 20));       // 8 MB gate preact
  float*    Rt = (float*)(ws + (20u << 20));       // 8 MB retention out
  float*    Zb = (float*)(ws + (28u << 20));       // 8 MB silu out (36 MB total)

  dim3 gg(BT / 64, DMODEL / 64);
  gemm512<1><<<gg, 256, 0, stream>>>(x, WQ, Qh, 0.125f);
  gemm512<1><<<gg, 256, 0, stream>>>(x, WK, Kh, 1.0f);
  gemm512<2><<<gg, 256, 0, stream>>>(x, WV, Vt, 1.0f);
  gemm512<0><<<gg, 256, 0, stream>>>(x, WG, Gp, 1.0f);

  retention_kernel<<<dim3(TSEQ / 64, NHEADS, 2), 128, 0, stream>>>(Qh, Kh, Vt,
                                                                   Rt);
  gn_gate_kernel<<<BT * NHEADS, 64, 0, stream>>>(Rt, Gp, gw, gb, Zb);

  gemm512<0><<<gg, 256, 0, stream>>>(Zb, WO, (float*)d_out, 1.0f);
}
